// BotRGCN_5901285065196
// MI455X (gfx1250) — compile-verified, matrix operand order checked by reading, not run on hardware
//
#include <hip/hip_runtime.h>
#include <hip/hip_bf16.h>

typedef __attribute__((ext_vector_type(16))) __bf16 v16bf;
typedef __attribute__((ext_vector_type(8)))  float  v8f;

#define LRELU_SLOPE 0.01f

// ---------------------------------------------------------------------------
// FAST PATH: requires M%16==0, K%32==0, Nc%16==0 (all heavy GEMMs qualify).
// Block = 256 threads = 8 waves computing a 128x16 C panel (waves stacked in M,
// all sharing one 16-column B panel). Per 32-K chunk the block stages B into
// LDS as split-bf16 (hi/lo) in exact WMMA B-operand order, so each lane reads
// its v16bf operand as one contiguous 32B ds access. No divergent guards ->
// no exec-mask branching around the WMMAs.
//
// Split-bf16: A=Ah+Al, B=Bh+Bl; D += Ah*Bh + Ah*Bl + Al*Bh  (~fp32 accuracy,
// f32 accumulate in the matrix core).
// ---------------------------------------------------------------------------
__global__ __launch_bounds__(256)
void gemm_wmma_fast(const float* __restrict__ A, int lda,
                    const float* __restrict__ B, int ldb,
                    const float* __restrict__ bias,
                    float* __restrict__ C, int ldc, int colOff,
                    int M, int K, int leaky)
{
    __shared__ __align__(32) __bf16 sbh[32][16];   // [lane][element] hi
    __shared__ __align__(32) __bf16 sbl[32][16];   // [lane][element] lo

    const int lane = threadIdx.x & 31;
    const int wave = threadIdx.x >> 5;
    const int tn   = blockIdx.y << 4;              // B/C column-tile base
    const int mt   = M >> 4;
    const int mtile = blockIdx.x * 8 + wave;
    const bool mvalid = (mtile < mt);              // wave-uniform
    const int tm = (mvalid ? mtile : (mt - 1)) << 4;  // clamp: compute, don't store

    const int hlf = lane >> 4;
    const int l16 = lane & 15;
    const long arowb = (long)(tm + l16) * lda;

    // B-staging decomposition: thread handles lane-slot ls, element pair j0,j0+1
    const int ls   = threadIdx.x >> 3;             // 0..31
    const int j0   = (threadIdx.x & 7) * 2;        // 0..14
    const int shlf = ls >> 4;
    const int scol = ls & 15;

    v8f acc = {0.f, 0.f, 0.f, 0.f, 0.f, 0.f, 0.f, 0.f};

    for (int k0 = 0; k0 < K; k0 += 32) {
        __syncthreads();                           // protect prior-iter LDS reads
        #pragma unroll
        for (int jj = 0; jj < 2; ++jj) {
            const int j  = j0 + jj;
            // operand element j of lane-half shlf covers K = 8*shlf + {j | 16+(j-8)}
            const int kk = k0 + 8 * shlf + ((j < 8) ? j : (16 + (j - 8)));
            const float bvf = B[(long)kk * ldb + tn + scol];
            const __bf16 h = (__bf16)bvf;
            sbh[ls][j] = h;
            sbl[ls][j] = (__bf16)(bvf - (float)h);
        }
        __syncthreads();

        // A: two contiguous 8-float runs per lane -> 4x b128 loads
        const float4* ap0 = (const float4*)(A + arowb + k0 + 8 * hlf);
        const float4* ap1 = (const float4*)(A + arowb + k0 + 16 + 8 * hlf);
        const float4 a0 = ap0[0], a1 = ap0[1], a2 = ap1[0], a3 = ap1[1];
        float af[16] = {a0.x,a0.y,a0.z,a0.w, a1.x,a1.y,a1.z,a1.w,
                        a2.x,a2.y,a2.z,a2.w, a3.x,a3.y,a3.z,a3.w};
        v16bf ah, al;
        #pragma unroll
        for (int j = 0; j < 16; ++j) {
            const __bf16 h = (__bf16)af[j];
            ah[j] = h;
            al[j] = (__bf16)(af[j] - (float)h);
        }

        const v16bf bh = *(const v16bf*)&sbh[lane][0];
        const v16bf bl = *(const v16bf*)&sbl[lane][0];

        acc = __builtin_amdgcn_wmma_f32_16x16x32_bf16(false, ah, false, bh,
                                                      (short)0, acc, false, false);
        acc = __builtin_amdgcn_wmma_f32_16x16x32_bf16(false, ah, false, bl,
                                                      (short)0, acc, false, false);
        acc = __builtin_amdgcn_wmma_f32_16x16x32_bf16(false, al, false, bh,
                                                      (short)0, acc, false, false);
    }

    if (mvalid) {                                  // wave-uniform
        const float bv = (bias != nullptr) ? bias[tn + l16] : 0.f;
        #pragma unroll
        for (int r = 0; r < 8; ++r) {
            const int m = tm + r + 8 * hlf;        // D: VGPR r -> row r (+8 hi half)
            float v = acc[r] + bv;
            if (leaky) v = (v > 0.f) ? v : LRELU_SLOPE * v;
            C[(long)m * ldc + colOff + tn + l16] = v;
        }
    }
}

// ---------------------------------------------------------------------------
// GENERIC PATH (odd K / narrow Nc: the K=5, K=3 encoders and the Nc=2 head).
// Guards are clamped-address loads + selects (no exec-mask branching).
// One wave per 16x16 tile.
// ---------------------------------------------------------------------------
__global__ __launch_bounds__(256)
void gemm_wmma_generic(const float* __restrict__ A, int lda,
                       const float* __restrict__ B, int ldb,
                       const float* __restrict__ bias,
                       float* __restrict__ C, int ldc, int colOff,
                       int M, int Nc, int K, int leaky)
{
    const int lane = threadIdx.x & 31;
    const int wave = threadIdx.x >> 5;
    const int mt = (M + 15) >> 4;
    const int nt = (Nc + 15) >> 4;
    const int tile = blockIdx.x * (blockDim.x >> 5) + wave;
    if (tile >= mt * nt) return;                   // wave-uniform exit

    const int tm  = (tile / nt) << 4;
    const int tn  = (tile % nt) << 4;
    const int hlf = lane >> 4;
    const int l16 = lane & 15;
    const int arow = tm + l16;
    const int bcol = tn + l16;
    const bool arv = (arow < M);
    const bool bcv = (bcol < Nc);
    const int arows = arv ? arow : (M - 1);        // clamped, always in-bounds
    const int bcols = bcv ? bcol : (Nc - 1);

    v8f acc = {0.f, 0.f, 0.f, 0.f, 0.f, 0.f, 0.f, 0.f};

    for (int k0 = 0; k0 < K; k0 += 32) {
        v16bf ah, al, bh, bl;
        #pragma unroll
        for (int j = 0; j < 16; ++j) {
            const int kk = k0 + 8 * hlf + ((j < 8) ? j : (16 + (j - 8)));
            const int kks = (kk < K) ? kk : (K - 1);
            const float ar = A[(long)arows * lda + kks];   // unconditional, clamped
            const float br = B[(long)kks * ldb + bcols];
            const float av = (arv && kk < K) ? ar : 0.f;   // v_cndmask select
            const float bv = (bcv && kk < K) ? br : 0.f;
            const __bf16 avh = (__bf16)av;
            const __bf16 bvh = (__bf16)bv;
            ah[j] = avh; al[j] = (__bf16)(av - (float)avh);
            bh[j] = bvh; bl[j] = (__bf16)(bv - (float)bvh);
        }
        acc = __builtin_amdgcn_wmma_f32_16x16x32_bf16(false, ah, false, bh,
                                                      (short)0, acc, false, false);
        acc = __builtin_amdgcn_wmma_f32_16x16x32_bf16(false, ah, false, bl,
                                                      (short)0, acc, false, false);
        acc = __builtin_amdgcn_wmma_f32_16x16x32_bf16(false, al, false, bh,
                                                      (short)0, acc, false, false);
    }

    const float bv = (bias != nullptr && bcv) ? bias[bcols] : 0.f;
    #pragma unroll
    for (int r = 0; r < 8; ++r) {
        const int m = tm + r + 8 * hlf;
        if (m < M && bcv) {
            float v = acc[r] + bv;
            if (leaky) v = (v > 0.f) ? v : LRELU_SLOPE * v;
            C[(long)m * ldc + colOff + bcol] = v;
        }
    }
}

// ---------------------------------------------------------------------------
// One wave per edge: float4 gather of h[rel][src], global_atomic_add_f32
// scatter into acc[rel][dst]; lane0 bumps the per-relation in-degree.
// ---------------------------------------------------------------------------
__global__ __launch_bounds__(256)
void rgcn_scatter(const int* __restrict__ ei, const int* __restrict__ et,
                  const float* __restrict__ h0, const float* __restrict__ h1,
                  float* __restrict__ acc0, float* __restrict__ acc1,
                  float* __restrict__ cnt0, float* __restrict__ cnt1, int E)
{
    const int e = (int)((blockIdx.x * (long)blockDim.x + threadIdx.x) >> 5);
    const int lane = threadIdx.x & 31;
    if (e >= E) return;
    const int src = ei[e];
    const int dst = ei[E + e];
    const int t   = et[e];
    const float4* h = (const float4*)(t ? h1 : h0);
    float* acc = t ? acc1 : acc0;
    #pragma unroll
    for (int i = 0; i < 2; ++i) {
        const int idx = i * 32 + lane;
        float4 v = h[(long)src * 64 + idx];
        float* p = acc + (long)dst * 256 + idx * 4;
        atomicAdd(p + 0, v.x);
        atomicAdd(p + 1, v.y);
        atomicAdd(p + 2, v.z);
        atomicAdd(p + 3, v.w);
    }
    if (lane == 0) atomicAdd(t ? (cnt1 + dst) : (cnt0 + dst), 1.0f);
}

__global__ __launch_bounds__(256)
void rgcn_finalize(float* __restrict__ out,
                   const float* __restrict__ acc0, const float* __restrict__ acc1,
                   const float* __restrict__ cnt0, const float* __restrict__ cnt1,
                   long total)
{
    const long i = (long)blockIdx.x * blockDim.x + threadIdx.x;
    if (i >= total) return;
    const int node = (int)(i >> 8);
    const float c0 = fmaxf(cnt0[node], 1.0f);
    const float c1 = fmaxf(cnt1[node], 1.0f);
    out[i] += acc0[i] / c0 + acc1[i] / c1;
}

__global__ __launch_bounds__(256)
void fill_zero(float* __restrict__ p, long n)
{
    const long i = (long)blockIdx.x * blockDim.x + threadIdx.x;
    if (i < n) p[i] = 0.f;
}

// ---------------------------------------------------------------------------
// Host-side orchestration
// ---------------------------------------------------------------------------
static inline void launch_gemm(const float* A, int lda, const float* B, int ldb,
                               const float* bias, float* C, int ldc, int colOff,
                               int M, int Nc, int K, int leaky, hipStream_t stream)
{
    if ((M & 15) == 0 && (K & 31) == 0 && (Nc & 15) == 0) {
        const int mt = M >> 4;
        dim3 grid((mt + 7) / 8, Nc >> 4);
        gemm_wmma_fast<<<grid, 256, 0, stream>>>(A, lda, B, ldb, bias, C, ldc,
                                                 colOff, M, K, leaky);
    } else {
        const int mt = (M + 15) / 16, nt = (Nc + 15) / 16;
        const int blocks = (mt * nt + 7) / 8;
        gemm_wmma_generic<<<blocks, 256, 0, stream>>>(A, lda, B, ldb, bias, C, ldc,
                                                      colOff, M, Nc, K, leaky);
    }
}

extern "C" void kernel_launch(void* const* d_in, const int* in_sizes, int n_in,
                              void* d_out, int out_size, void* d_ws, size_t ws_size,
                              hipStream_t stream)
{
    const float* des      = (const float*)d_in[0];
    const float* tweet    = (const float*)d_in[1];
    const float* num_prop = (const float*)d_in[2];
    const float* cat_prop = (const float*)d_in[3];
    const int*   ei       = (const int*)d_in[4];
    const int*   et       = (const int*)d_in[5];
    const float* Wd = (const float*)d_in[6],  *bd = (const float*)d_in[7];
    const float* Wt = (const float*)d_in[8],  *bt = (const float*)d_in[9];
    const float* Wn = (const float*)d_in[10], *bn = (const float*)d_in[11];
    const float* Wc = (const float*)d_in[12], *bc = (const float*)d_in[13];
    const float* Wi = (const float*)d_in[14], *bi = (const float*)d_in[15];
    const float* rel_w1  = (const float*)d_in[16];
    const float* root_w1 = (const float*)d_in[17];
    const float* bias1   = (const float*)d_in[18];
    const float* rel_w2  = (const float*)d_in[19];
    const float* root_w2 = (const float*)d_in[20];
    const float* bias2   = (const float*)d_in[21];
    const float* Wo1 = (const float*)d_in[22], *bo1 = (const float*)d_in[23];
    const float* Wo2 = (const float*)d_in[24], *bo2 = (const float*)d_in[25];
    float* out = (float*)d_out;

    const int N = in_sizes[0] / 768;
    const int E = in_sizes[5];
    const int H = 256;
    const long NH = (long)N * H;

    // workspace layout (floats): xA | xB | h0 | h1 | acc0 | acc1 | cnt0 | cnt1
    float* xA   = (float*)d_ws;
    float* xB   = xA + NH;
    float* h0   = xB + NH;
    float* h1   = h0 + NH;
    float* acc0 = h1 + NH;
    float* acc1 = acc0 + NH;
    float* cnt0 = acc1 + NH;
    float* cnt1 = cnt0 + N;

    const long accRegion = 2 * NH + 2 * N;   // acc0..cnt1 contiguous
    const int  fillBlocks = (int)((accRegion + 255) / 256);
    const int  scatBlocks = (E * 32 + 255) / 256;
    const int  finBlocks  = (int)((NH + 255) / 256);

    // 1) feature encoders -> concat into xA  [N,256], fused LeakyReLU
    launch_gemm(des,      768, Wd, 64, bd, xA, H,   0, N,  64, 768, 1, stream);
    launch_gemm(tweet,    768, Wt, 64, bt, xA, H,  64, N,  64, 768, 1, stream);
    launch_gemm(num_prop,   5, Wn, 64, bn, xA, H, 128, N,  64,   5, 1, stream);
    launch_gemm(cat_prop,   3, Wc, 64, bc, xA, H, 192, N,  64,   3, 1, stream);

    // 2) xB = lrelu(xA @ Wi + bi)
    launch_gemm(xA, H, Wi, H, bi, xB, H, 0, N, H, H, 1, stream);

    // 3) RGCN layer 1: x = xB -> xA
    fill_zero<<<fillBlocks, 256, 0, stream>>>(acc0, accRegion);
    launch_gemm(xB, H, root_w1,        H, bias1,   xA, H, 0, N, H, H, 0, stream);
    launch_gemm(xB, H, rel_w1,         H, nullptr, h0, H, 0, N, H, H, 0, stream);
    launch_gemm(xB, H, rel_w1 + H * H, H, nullptr, h1, H, 0, N, H, H, 0, stream);
    rgcn_scatter<<<scatBlocks, 256, 0, stream>>>(ei, et, h0, h1, acc0, acc1, cnt0, cnt1, E);
    rgcn_finalize<<<finBlocks, 256, 0, stream>>>(xA, acc0, acc1, cnt0, cnt1, NH);

    // 4) RGCN layer 2: x = xA -> xB
    fill_zero<<<fillBlocks, 256, 0, stream>>>(acc0, accRegion);
    launch_gemm(xA, H, root_w2,        H, bias2,   xB, H, 0, N, H, H, 0, stream);
    launch_gemm(xA, H, rel_w2,         H, nullptr, h0, H, 0, N, H, H, 0, stream);
    launch_gemm(xA, H, rel_w2 + H * H, H, nullptr, h1, H, 0, N, H, H, 0, stream);
    rgcn_scatter<<<scatBlocks, 256, 0, stream>>>(ei, et, h0, h1, acc0, acc1, cnt0, cnt1, E);
    rgcn_finalize<<<finBlocks, 256, 0, stream>>>(xB, acc0, acc1, cnt0, cnt1, NH);

    // 5) xA = lrelu(xB @ Wo1 + bo1)
    launch_gemm(xB, H, Wo1, H, bo1, xA, H, 0, N, H, H, 1, stream);

    // 6) out = xA @ Wo2 + bo2   [N,2]
    launch_gemm(xA, H, Wo2, 2, bo2, out, 2, 0, N, 2, H, 0, stream);
}
